// GraphProp_66992899883697
// MI455X (gfx1250) — matrix operand order repaired and checked by reading, not compiled, not run on hardware
//
#include <hip/hip_runtime.h>
#include <stdint.h>

#define H   64
#define H2  128
#define H3  192
#define WMSG_K 129
#define WMSG_STRIDE 144   // padded bf16 K-stride (mult of 16 -> 32B rows, +8 banks skew)
#define WIH_STRIDE  144
#define WHH_STRIDE  80

typedef __attribute__((ext_vector_type(16))) __bf16 v16bf;
typedef __attribute__((ext_vector_type(8)))  float  v8f;

union Frag16 { v16bf v; uint4 q[2]; unsigned short s[16]; };

__device__ __forceinline__ unsigned short f2bf(float f) {
  unsigned u = __float_as_uint(f);
  unsigned r = u + 0x7FFFu + ((u >> 16) & 1u);   // round-to-nearest-even
  return (unsigned short)(r >> 16);
}
__device__ __forceinline__ float sigmoidf_(float x) { return 1.0f / (1.0f + __expf(-x)); }

// CDNA5 async global->LDS copy (ASYNCcnt-tracked), 16B per lane per issue.
__device__ __forceinline__ void async_g2l_b128(void* lds, const void* gmem) {
  unsigned dsoff = (unsigned)(unsigned long long)lds;   // LDS aperture: low 32 bits = LDS offset
  asm volatile("global_load_async_to_lds_b128 %0, %1, off"
               :: "v"(dsoff), "v"(gmem) : "memory");
}
__device__ __forceinline__ void wait_async0() {
  asm volatile("s_wait_asynccnt 0" ::: "memory");
}

// ---------------- weight conversion: fp32 -> padded bf16 ----------------
__global__ void prep_weights_kernel(const float* __restrict__ Wmsg,
                                    const float* __restrict__ Wih,
                                    const float* __restrict__ Whh,
                                    unsigned short* __restrict__ WmsgBf,
                                    unsigned short* __restrict__ WihBf,
                                    unsigned short* __restrict__ WhhBf, int T) {
  const long tot0 = (long)T * H2 * WMSG_STRIDE;
  const long tot1 = (long)T * H3 * WIH_STRIDE;
  const long tot2 = (long)T * H3 * WHH_STRIDE;
  const long total = tot0 + tot1 + tot2;
  for (long i = (long)blockIdx.x * blockDim.x + threadIdx.x; i < total;
       i += (long)gridDim.x * blockDim.x) {
    if (i < tot0) {
      long t = i / (H2 * WMSG_STRIDE), r = (i / WMSG_STRIDE) % H2, k = i % WMSG_STRIDE;
      WmsgBf[i] = (k < H2) ? f2bf(Wmsg[(t * H2 + r) * WMSG_K + k]) : (unsigned short)0;
    } else if (i < tot0 + tot1) {
      long j = i - tot0;
      long t = j / (H3 * WIH_STRIDE), r = (j / WIH_STRIDE) % H3, k = j % WIH_STRIDE;
      WihBf[j] = (k < H2) ? f2bf(Wih[(t * H3 + r) * H2 + k]) : (unsigned short)0;
    } else {
      long j = i - tot0 - tot1;
      long t = j / (H3 * WHH_STRIDE), r = (j / WHH_STRIDE) % H3, k = j % WHH_STRIDE;
      WhhBf[j] = (k < H) ? f2bf(Whh[(t * H3 + r) * H + k]) : (unsigned short)0;
    }
  }
}

// ---------------- per-round prep: hv -> bf16, zero accumulator ----------------
__global__ void prep_round_kernel(const float* __restrict__ hv,
                                  unsigned short* __restrict__ hvbf,
                                  float* __restrict__ a, long N) {
  const long n64 = N * H, n128 = N * H2;
  for (long i = (long)blockIdx.x * blockDim.x + threadIdx.x; i < n128;
       i += (long)gridDim.x * blockDim.x) {
    if (i < n64) hvbf[i] = f2bf(hv[i]);
    a[i] = 0.0f;
  }
}

// ---------------- edge messages: [16 x 129] @ W^T -> atomic scatter into a ----------------
// dyn LDS: 128*144 bf16 W tile + 128 f32 Wlast + 128 f32 bias  (37.9 KB)
__global__ void edge_msg_kernel(const unsigned short* __restrict__ hvbf,
                                const int* __restrict__ src, const int* __restrict__ dst,
                                const float* __restrict__ he,
                                const float* __restrict__ Wmsg,  // fp32, k=128 column
                                const float* __restrict__ bmsg,
                                const unsigned short* __restrict__ WmsgBf,
                                float* __restrict__ a, int t, int numTiles) {
  extern __shared__ __align__(16) unsigned char smem_raw[];
  unsigned short* sW   = (unsigned short*)smem_raw;                  // [128][144]
  float* sWlast        = (float*)(smem_raw + H2 * WMSG_STRIDE * 2);  // [128]
  float* sBias         = sWlast + H2;                                 // [128]

  const int tid = threadIdx.x;
  {
    const unsigned short* wsrc = WmsgBf + (size_t)t * H2 * WMSG_STRIDE;
    const int chunks = (H2 * WMSG_STRIDE) / 8;       // 2304, divides 128 exactly
    for (int i = tid; i < chunks; i += blockDim.x)
      async_g2l_b128(sW + i * 8, wsrc + i * 8);
    for (int i = tid; i < H2; i += blockDim.x) {
      sWlast[i] = Wmsg[((size_t)t * H2 + i) * WMSG_K + H2];
      sBias[i]  = bmsg[t * H2 + i];
    }
    wait_async0();
    __syncthreads();
  }

  const int lane = tid & 31;
  const int wave = tid >> 5;
  const int wavesPerGrid = (blockDim.x >> 5) * gridDim.x;
  const int row  = lane & 15;
  const bool hi  = lane >= 16;
  const int off8 = hi ? 8 : 0;
  const unsigned short* wrow0 = sW + (size_t)row * WMSG_STRIDE + (hi ? 8 : 0);

  for (int tile = blockIdx.x * (blockDim.x >> 5) + wave; tile < numTiles;
       tile += wavesPerGrid) {
    asm volatile("" ::: "memory");   // keep B reads in-loop (ds_load, not spilled regs)
    const int e0 = tile * 16;
    const int eA = e0 + row;
    const int dA = dst[eA];
    const int sA = src[eA];

    // A fragments (16x32 bf16 layout): K 0..63 = h_dst, 64..127 = h_src
    Frag16 A[4];
#pragma unroll
    for (int ks = 0; ks < 4; ++ks) {
      const int k0 = ks * 32;
      const unsigned short* base =
          hvbf + (size_t)(k0 < H ? dA : sA) * H + (k0 & (H - 1)) + off8;
      A[ks].q[0] = *(const uint4*)(base);
      A[ks].q[1] = *(const uint4*)(base + 16);
    }

    float heR[8]; float* paR[8];
#pragma unroll
    for (int r = 0; r < 8; ++r) {
      const int e = e0 + r + (hi ? 8 : 0);
      heR[r] = he[e];
      paR[r] = a + (size_t)dst[e] * H2;
    }

#pragma unroll
    for (int nt = 0; nt < 8; ++nt) {
      const int j = nt * 16 + row;                 // output column / W row
      const unsigned short* wrow = wrow0 + (size_t)nt * 16 * WMSG_STRIDE;
      v8f C = {0.f, 0.f, 0.f, 0.f, 0.f, 0.f, 0.f, 0.f};
#pragma unroll
      for (int ks = 0; ks < 4; ++ks) {
        Frag16 B;
        B.q[0] = *(const uint4*)(wrow + ks * 32);
        B.q[1] = *(const uint4*)(wrow + ks * 32 + 16);
        C = __builtin_amdgcn_wmma_f32_16x16x32_bf16(false, A[ks].v, false, B.v,
                                                    (short)0, C, false, false);
      }
      const float wl = sWlast[j];
      const float bj = sBias[j];
#pragma unroll
      for (int r = 0; r < 8; ++r) {
        atomicAdd(paR[r] + j, C[r] + heR[r] * wl + bj);
      }
    }
  }
}

// ---------------- GRU update per 16-node tile ----------------
// dyn LDS: 192*144 bf16 Wih + 192*80 bf16 Whh + 2*192 f32 biases (87.6 KB)
__global__ void gru_kernel(const float* __restrict__ a,
                           const unsigned short* __restrict__ hvbf,
                           const float* __restrict__ hv_cur,
                           const unsigned short* __restrict__ WihBf,
                           const unsigned short* __restrict__ WhhBf,
                           const float* __restrict__ bih, const float* __restrict__ bhh,
                           float* __restrict__ hv_next, int t, int numTiles) {
  extern __shared__ __align__(16) unsigned char smem_raw[];
  unsigned short* sWih = (unsigned short*)smem_raw;                              // [192][144]
  unsigned short* sWhh = sWih + H3 * WIH_STRIDE;                                 // [192][80]
  float* sBih = (float*)(smem_raw + (H3 * WIH_STRIDE + H3 * WHH_STRIDE) * 2);    // [192]
  float* sBhh = sBih + H3;                                                        // [192]

  const int tid = threadIdx.x;
  {
    const unsigned short* wi = WihBf + (size_t)t * H3 * WIH_STRIDE;
    const unsigned short* wh = WhhBf + (size_t)t * H3 * WHH_STRIDE;
    const int c0 = (H3 * WIH_STRIDE) / 8;   // 3456
    const int c1 = (H3 * WHH_STRIDE) / 8;   // 1920
    for (int i = tid; i < c0; i += blockDim.x) async_g2l_b128(sWih + i * 8, wi + i * 8);
    for (int i = tid; i < c1; i += blockDim.x) async_g2l_b128(sWhh + i * 8, wh + i * 8);
    for (int i = tid; i < H3; i += blockDim.x) {
      sBih[i] = bih[t * H3 + i];
      sBhh[i] = bhh[t * H3 + i];
    }
    wait_async0();
    __syncthreads();
  }

  const int lane = tid & 31;
  const int wave = tid >> 5;
  const int wavesPerGrid = (blockDim.x >> 5) * gridDim.x;
  const int row  = lane & 15;
  const bool hi  = lane >= 16;
  const int off8 = hi ? 8 : 0;

  for (int tile = blockIdx.x * (blockDim.x >> 5) + wave; tile < numTiles;
       tile += wavesPerGrid) {
    asm volatile("" ::: "memory");
    const int n0 = tile * 16;

    // A fragments of `a` (fp32 -> bf16 in-register), K = 128
    const float* arow = a + (size_t)(n0 + row) * H2;
    Frag16 Aa[4];
#pragma unroll
    for (int ks = 0; ks < 4; ++ks) {
      const float* p = arow + ks * 32 + off8;
#pragma unroll
      for (int q = 0; q < 2; ++q) {
        float4 f0 = *(const float4*)(p + q * 16);
        float4 f1 = *(const float4*)(p + q * 16 + 4);
        const int b = q * 8;
        Aa[ks].s[b + 0] = f2bf(f0.x); Aa[ks].s[b + 1] = f2bf(f0.y);
        Aa[ks].s[b + 2] = f2bf(f0.z); Aa[ks].s[b + 3] = f2bf(f0.w);
        Aa[ks].s[b + 4] = f2bf(f1.x); Aa[ks].s[b + 5] = f2bf(f1.y);
        Aa[ks].s[b + 6] = f2bf(f1.z); Aa[ks].s[b + 7] = f2bf(f1.w);
      }
    }
    // A fragments of h (bf16 already), K = 64
    Frag16 Ah[2];
#pragma unroll
    for (int ks = 0; ks < 2; ++ks) {
      const unsigned short* p = hvbf + (size_t)(n0 + row) * H + ks * 32 + off8;
      Ah[ks].q[0] = *(const uint4*)(p);
      Ah[ks].q[1] = *(const uint4*)(p + 16);
    }

#pragma unroll
    for (int g = 0; g < 4; ++g) {     // 16 output columns per group
      v8f Cgi[3], Cgh[3];
      float bi[3], bh[3];
#pragma unroll
      for (int s3 = 0; s3 < 3; ++s3) {  // r / z / n gate sections
        const int jj = s3 * H + g * 16 + row;
        bi[s3] = sBih[jj];
        bh[s3] = sBhh[jj];

        v8f Ci = {0.f, 0.f, 0.f, 0.f, 0.f, 0.f, 0.f, 0.f};
        const unsigned short* wr = sWih + (size_t)jj * WIH_STRIDE + (hi ? 8 : 0);
#pragma unroll
        for (int ks = 0; ks < 4; ++ks) {
          Frag16 B;
          B.q[0] = *(const uint4*)(wr + ks * 32);
          B.q[1] = *(const uint4*)(wr + ks * 32 + 16);
          Ci = __builtin_amdgcn_wmma_f32_16x16x32_bf16(false, Aa[ks].v, false, B.v,
                                                       (short)0, Ci, false, false);
        }
        Cgi[s3] = Ci;

        v8f Ch = {0.f, 0.f, 0.f, 0.f, 0.f, 0.f, 0.f, 0.f};
        const unsigned short* wr2 = sWhh + (size_t)jj * WHH_STRIDE + (hi ? 8 : 0);
#pragma unroll
        for (int ks = 0; ks < 2; ++ks) {
          Frag16 B;
          B.q[0] = *(const uint4*)(wr2 + ks * 32);
          B.q[1] = *(const uint4*)(wr2 + ks * 32 + 16);
          Ch = __builtin_amdgcn_wmma_f32_16x16x32_bf16(false, Ah[ks].v, false, B.v,
                                                       (short)0, Ch, false, false);
        }
        Cgh[s3] = Ch;
      }

      const int j = g * 16 + row;
#pragma unroll
      for (int r = 0; r < 8; ++r) {
        const int node = n0 + r + (hi ? 8 : 0);
        const float hp  = hv_cur[(size_t)node * H + j];
        const float ir  = Cgi[0][r] + bi[0], iz = Cgi[1][r] + bi[1], in_ = Cgi[2][r] + bi[2];
        const float hr  = Cgh[0][r] + bh[0], hz = Cgh[1][r] + bh[1], hn  = Cgh[2][r] + bh[2];
        const float rg = sigmoidf_(ir + hr);
        const float zg = sigmoidf_(iz + hz);
        const float ng = tanhf(in_ + rg * hn);
        hv_next[(size_t)node * H + j] = (1.0f - zg) * ng + zg * hp;
      }
    }
  }
}

extern "C" void kernel_launch(void* const* d_in, const int* in_sizes, int n_in,
                              void* d_out, int out_size, void* d_ws, size_t ws_size,
                              hipStream_t stream) {
  const float* hv_in = (const float*)d_in[0];
  const float* he    = (const float*)d_in[1];
  const int*   src   = (const int*)d_in[2];
  const int*   dst   = (const int*)d_in[3];
  const float* Wmsg  = (const float*)d_in[4];
  const float* bmsg  = (const float*)d_in[5];
  const float* Wih   = (const float*)d_in[6];
  const float* Whh   = (const float*)d_in[7];
  const float* bih   = (const float*)d_in[8];
  const float* bhh   = (const float*)d_in[9];

  const long N = in_sizes[0] / H;
  const int  E = in_sizes[2];
  const int  T = in_sizes[4] / (H2 * WMSG_K);

  char* ws = (char*)d_ws;
  size_t off = 0;
  auto take = [&](size_t bytes) -> char* {
    char* p = ws + off;
    off += (bytes + 255) & ~(size_t)255;
    return p;
  };
  float* bufA = (float*)take((size_t)N * H * 4);
  float* bufB = (float*)take((size_t)N * H * 4);
  unsigned short* hvbf = (unsigned short*)take((size_t)N * H * 2);
  float* aacc = (float*)take((size_t)N * H2 * 4);
  unsigned short* WmsgBf = (unsigned short*)take((size_t)T * H2 * WMSG_STRIDE * 2);
  unsigned short* WihBf  = (unsigned short*)take((size_t)T * H3 * WIH_STRIDE * 2);
  unsigned short* WhhBf  = (unsigned short*)take((size_t)T * H3 * WHH_STRIDE * 2);

  prep_weights_kernel<<<128, 256, 0, stream>>>(Wmsg, Wih, Whh, WmsgBf, WihBf, WhhBf, T);

  const size_t edgeSmem = (size_t)H2 * WMSG_STRIDE * 2 + (size_t)H2 * 4 * 2;
  const size_t gruSmem  = (size_t)(H3 * WIH_STRIDE + H3 * WHH_STRIDE) * 2 + (size_t)H3 * 4 * 2;

  const int edgeTiles = E / 16;
  const int nodeTiles = (int)(N / 16);
  for (int t = 0; t < T; ++t) {
    const float* hv_cur = (t == 0) ? hv_in : ((t & 1) ? bufA : bufB);
    float* hv_next = (t == T - 1) ? (float*)d_out : ((t & 1) ? bufB : bufA);

    prep_round_kernel<<<2048, 256, 0, stream>>>(hv_cur, hvbf, aacc, N);
    edge_msg_kernel<<<1024, 128, edgeSmem, stream>>>(hvbf, src, dst, he, Wmsg, bmsg,
                                                     WmsgBf, aacc, t, edgeTiles);
    gru_kernel<<<512, 128, gruSmem, stream>>>(aacc, hvbf, hv_cur, WihBf, WhhBf,
                                              bih, bhh, hv_next, t, nodeTiles);
  }
}
